// GATModel_69355131896599
// MI455X (gfx1250) — matrix operand order repaired and checked by reading, not compiled, not run on hardware
//
#include <hip/hip_runtime.h>
#include <hip/hip_bf16.h>

typedef __attribute__((ext_vector_type(16))) __bf16 v16bf;
typedef __attribute__((ext_vector_type(8)))  float  v8f;

union BF16x2 { unsigned int u; __bf16 h[2]; };
union BF16U  { __bf16 h; unsigned short s; };
union AFrag  { unsigned int u[8]; v16bf v; };
union CFrag  { float f[8]; v8f v; };

__device__ __forceinline__ unsigned short f2bf(float f) { BF16U t; t.h = (__bf16)f; return t.s; }
__device__ __forceinline__ float sigf(float x)  { return 1.0f / (1.0f + __expf(-x)); }
__device__ __forceinline__ float leakyf(float x){ return x > 0.0f ? x : 0.01f * x; }

// ---------------------------------------------------------------------------
// Persistent LSTM layer kernel. One block = 64 rows of N, loops over T.
// gates(64x256) = x_t @ Wih^T + h @ Whh^T + b via v_wmma_f32_16x16x32_bf16.
// Whh B-fragments for the first 4 of 8 tiles are hoisted into registers
// (64 VGPRs -- stays under the 256-VGPR architected window, avoiding
// s_set_vgpr_msb churn); the other 4 tiles stream B from LDS, with their
// latency hidden behind the register-resident WMMA chain.
// IN64==0: input is (N,T,6) fp32, projected with scalar FMAs into C-fragments.
// IN64==1: input is (N,T,64) bf16, projected with WMMA (B-frags from LDS).
// h kept in LDS (bf16), c kept in registers (16 values per thread).
// ---------------------------------------------------------------------------
template<int IN64, int STORE_SEQ, int STORE_LAST>
__global__ __launch_bounds__(256) void lstm_layer_kernel(
    const float* __restrict__ xin32,          // (N,T,6) fp32  (IN64==0)
    const unsigned int* __restrict__ xin16,   // (N,T,64) bf16 pairs (IN64==1)
    const float* __restrict__ Wih,            // (256, D)
    const float* __restrict__ Whh,            // (256, 64)
    const float* __restrict__ bih,
    const float* __restrict__ bhh,
    unsigned short* __restrict__ out_seq,     // (N,T,64) bf16
    float* __restrict__ out_last,             // (N,64) fp32
    int Tt)
{
    __shared__ unsigned int sWhh_u[256 * 32];                 // bf16 [256][64], 32 KB
    __shared__ unsigned int sWih_u[IN64 ? 256 * 32 : 4];      // bf16 [256][64] (layer1)
    __shared__ float        sWih_f[IN64 ? 4 : 256 * 6];       // fp32 [256][6]  (layer0)
    __shared__ float        sBias[256];
    __shared__ float        sGates[64 * 256];                 // 64 KB
    __shared__ unsigned int sH_u[64 * 32];                    // bf16 [64][64], 8 KB

    const int tid = threadIdx.x;

    for (int i = tid; i < 256 * 32; i += 256) {
        BF16x2 p; p.h[0] = (__bf16)Whh[2 * i]; p.h[1] = (__bf16)Whh[2 * i + 1];
        sWhh_u[i] = p.u;
    }
    if (IN64) {
        for (int i = tid; i < 256 * 32; i += 256) {
            BF16x2 p; p.h[0] = (__bf16)Wih[2 * i]; p.h[1] = (__bf16)Wih[2 * i + 1];
            sWih_u[i] = p.u;
        }
    } else {
        for (int i = tid; i < 256 * 6; i += 256) sWih_f[i] = Wih[i];
    }
    sBias[tid] = bih[tid] + bhh[tid];
    for (int i = tid; i < 64 * 32; i += 256) sH_u[i] = 0u;

    float creg[16];
    #pragma unroll
    for (int q = 0; q < 16; ++q) creg[q] = 0.0f;

    const int wave = tid >> 5, lane = tid & 31;
    const int l16  = lane & 15, lh = lane >> 4;
    const int mt   = wave >> 1;            // M-tile (0..3)
    const int nt0  = (wave & 1) * 8;       // first N-tile of this wave
    const int row0 = blockIdx.x * 64;
    const int grow  = tid >> 2;            // row for gate-nonlinearity phase
    const int gcol0 = (tid & 3) * 16;

    __syncthreads();

    // --- Hoist Whh B-fragments for tiles 0..3 into registers (4 x 2 x 8 VGPRs) ---
    AFrag bh[4][2];
    #pragma unroll
    for (int i = 0; i < 4; ++i) {
        const int g = (nt0 + i) * 16 + l16;
        #pragma unroll
        for (int kc = 0; kc < 2; ++kc) {
            const int ks = kc * 32 + lh * 16;
            #pragma unroll
            for (int v = 0; v < 8; ++v)
                bh[i][kc].u[v] = sWhh_u[(g * 64 + ks + 2 * v) >> 1];
        }
    }

    for (int t = 0; t < Tt; ++t) {
        // --- A fragments: h (from LDS) and x_t (from global, layer1 only) ---
        AFrag ah[2], ax[2];
        {
            const int m = mt * 16 + l16;
            if (IN64 && (t + 1 < Tt)) {
                // prefetch next timestep's input row into cache
                __builtin_prefetch(
                    &xin16[(((size_t)(row0 + m) * Tt + (t + 1)) * 64) >> 1], 0, 1);
            }
            #pragma unroll
            for (int kc = 0; kc < 2; ++kc) {
                const int kb = kc * 32 + lh * 16;
                #pragma unroll
                for (int v = 0; v < 8; ++v) {
                    const int k = kb + 2 * v;
                    ah[kc].u[v] = sH_u[(m * 64 + k) >> 1];
                    if (IN64)
                        ax[kc].u[v] = xin16[(((size_t)(row0 + m) * Tt + t) * 64 + k) >> 1];
                }
            }
        }
        // --- layer0: preload x rows for scalar input projection ---
        float xr[8][6];
        if (!IN64) {
            #pragma unroll
            for (int v = 0; v < 8; ++v) {
                const size_t base = ((size_t)(row0 + mt * 16 + lh * 8 + v) * Tt + t) * 6;
                #pragma unroll
                for (int d = 0; d < 6; ++d) xr[v][d] = xin32[base + d];
            }
        }
        // --- 8 output tiles per wave: 16x16 gates, K=64 ---
        #pragma unroll
        for (int i = 0; i < 8; ++i) {
            const int nt = nt0 + i;
            const int g  = nt * 16 + l16;
            CFrag acc;
            #pragma unroll
            for (int v = 0; v < 8; ++v) {
                float c0 = sBias[g];
                if (!IN64) {
                    const float* wp = &sWih_f[g * 6];
                    #pragma unroll
                    for (int d = 0; d < 6; ++d) c0 += xr[v][d] * wp[d];
                }
                acc.f[v] = c0;
            }
            // recurrent term: tiles 0..3 use register-resident B, 4..7 stream from LDS
            #pragma unroll
            for (int kc = 0; kc < 2; ++kc) {
                AFrag b;
                if (i < 4) {
                    b = bh[i][kc];
                } else {
                    const int ks = kc * 32 + lh * 16;
                    #pragma unroll
                    for (int v = 0; v < 8; ++v) b.u[v] = sWhh_u[(g * 64 + ks + 2 * v) >> 1];
                }
                acc.v = __builtin_amdgcn_wmma_f32_16x16x32_bf16(
                            false, ah[kc].v, false, b.v, (short)0, acc.v, false, false);
                if (IN64) {
                    AFrag bx;
                    const int ks = kc * 32 + lh * 16;
                    #pragma unroll
                    for (int v = 0; v < 8; ++v) bx.u[v] = sWih_u[(g * 64 + ks + 2 * v) >> 1];
                    acc.v = __builtin_amdgcn_wmma_f32_16x16x32_bf16(
                                false, ax[kc].v, false, bx.v, (short)0, acc.v, false, false);
                }
            }
            #pragma unroll
            for (int v = 0; v < 8; ++v)
                sGates[(mt * 16 + lh * 8 + v) * 256 + g] = acc.f[v];
        }
        __syncthreads();

        // --- LSTM cell nonlinearities; update c (regs) and h (LDS, bf16) ---
        #pragma unroll
        for (int q = 0; q < 16; ++q) {
            const int col = gcol0 + q;
            const float ig = sGates[grow * 256 + col];
            const float fg = sGates[grow * 256 + col + 64];
            const float gg = sGates[grow * 256 + col + 128];
            const float og = sGates[grow * 256 + col + 192];
            const float cn = sigf(fg) * creg[q] + sigf(ig) * tanhf(gg);
            creg[q] = cn;
            const float h = sigf(og) * tanhf(cn);
            ((unsigned short*)sH_u)[grow * 64 + col] = f2bf(h);
            if (STORE_SEQ)
                out_seq[((size_t)(row0 + grow) * Tt + t) * 64 + col] = f2bf(h);
            if (STORE_LAST && (t == Tt - 1))
                out_last[(size_t)(row0 + grow) * 64 + col] = h;
        }
        __syncthreads();
    }
}

// ---------------------------------------------------------------------------
// out(N x 64) = act(in(N x 64) @ W^T(64x64) + b) via WMMA bf16.
// Block = 128 rows, 8 waves: wave w owns M-tile w, loops 4 N-tiles, K=64.
// ---------------------------------------------------------------------------
template<int ACT>
__global__ __launch_bounds__(256) void gemm64_kernel(
    const float* __restrict__ in, const float* __restrict__ W,
    const float* __restrict__ bias, float* __restrict__ out)
{
    __shared__ unsigned int sW_u[64 * 32];  // bf16 [64][64]
    __shared__ float sB[64];
    const int tid = threadIdx.x;
    for (int i = tid; i < 64 * 32; i += 256) {
        BF16x2 p; p.h[0] = (__bf16)W[2 * i]; p.h[1] = (__bf16)W[2 * i + 1];
        sW_u[i] = p.u;
    }
    if (tid < 64) sB[tid] = bias[tid];
    __syncthreads();

    const int wave = tid >> 5, lane = tid & 31;
    const int l16 = lane & 15, lh = lane >> 4;
    const int row0 = blockIdx.x * 128;
    const int mt = wave;
    const int m  = row0 + mt * 16 + l16;

    AFrag a[2];
    #pragma unroll
    for (int kc = 0; kc < 2; ++kc) {
        const int kb = kc * 32 + lh * 16;
        #pragma unroll
        for (int v = 0; v < 8; ++v) {
            const int k = kb + 2 * v;
            BF16x2 p;
            p.h[0] = (__bf16)in[(size_t)m * 64 + k];
            p.h[1] = (__bf16)in[(size_t)m * 64 + k + 1];
            a[kc].u[v] = p.u;
        }
    }
    #pragma unroll
    for (int nt = 0; nt < 4; ++nt) {
        const int g = nt * 16 + l16;
        CFrag acc;
        #pragma unroll
        for (int v = 0; v < 8; ++v) acc.f[v] = sB[g];
        #pragma unroll
        for (int kc = 0; kc < 2; ++kc) {
            AFrag b;
            const int ks = kc * 32 + lh * 16;
            #pragma unroll
            for (int v = 0; v < 8; ++v) b.u[v] = sW_u[(g * 64 + ks + 2 * v) >> 1];
            acc.v = __builtin_amdgcn_wmma_f32_16x16x32_bf16(
                        false, a[kc].v, false, b.v, (short)0, acc.v, false, false);
        }
        #pragma unroll
        for (int v = 0; v < 8; ++v) {
            float r = acc.f[v];
            if (ACT) r = leakyf(r);
            out[(size_t)(row0 + mt * 16 + lh * 8 + v) * 64 + g] = r;
        }
    }
}

// s1[i] = h'[i] . a[0:64] ; s2[i] = h'[i] . a[64:128]
__global__ void gat_scores_kernel(const float* __restrict__ hp, const float* __restrict__ av,
                                  float* __restrict__ s1, float* __restrict__ s2)
{
    const int r = blockIdx.x * blockDim.x + threadIdx.x;
    float a1 = 0.0f, a2 = 0.0f;
    #pragma unroll 8
    for (int k = 0; k < 64; ++k) {
        const float h = hp[(size_t)r * 64 + k];
        a1 += h * av[k];
        a2 += h * av[64 + k];
    }
    s1[r] = a1; s2[r] = a2;
}

// Flash-style attention: res[i] = sum_j softmax_j(leaky(s2[i]+s1[j])) x[j] + x[i]
// 64 rows per block; 4 lanes per row, each owning 16 feature dims. The j-stream
// (last: 2 MB) is L2-resident; per-chunk tiles staged through LDS.
__global__ __launch_bounds__(256) void gat_attn_kernel(
    const float* __restrict__ xlast, const float* __restrict__ s1,
    const float* __restrict__ s2, float* __restrict__ res, int Nn)
{
    __shared__ float sX[256 * 64];
    __shared__ float sS1[256];
    const int tid = threadIdx.x;
    const int r   = blockIdx.x * 64 + (tid >> 2);
    const int d0  = (tid & 3) * 16;

    float acc[16];
    #pragma unroll
    for (int q = 0; q < 16; ++q) acc[q] = 0.0f;
    float mI = -1e30f, dI = 0.0f;
    const float s2i = s2[r];

    for (int jb = 0; jb < Nn; jb += 256) {
        __syncthreads();
        #pragma unroll
        for (int k = 0; k < 64; k += 4)
            *(float4*)&sX[tid * 64 + k] = *(const float4*)&xlast[(size_t)(jb + tid) * 64 + k];
        sS1[tid] = s1[jb + tid];
        __syncthreads();
        for (int jj = 0; jj < 256; ++jj) {
            float s = s2i + sS1[jj];
            s = s > 0.0f ? s : 0.01f * s;
            const float mN = fmaxf(mI, s);
            const float sc = __expf(mI - mN);
            const float w  = __expf(s - mN);
            dI = dI * sc + w;
            const float* xp = &sX[jj * 64 + d0];
            #pragma unroll
            for (int q = 0; q < 16; ++q) acc[q] = acc[q] * sc + w * xp[q];
            mI = mN;
        }
    }
    const float inv = 1.0f / dI;
    #pragma unroll
    for (int q = 0; q < 16; ++q)
        res[(size_t)r * 64 + d0 + q] = acc[q] * inv + xlast[(size_t)r * 64 + d0 + q];
}

// out[i] = hidden[i] . Wout + bout
__global__ void out_head_kernel(const float* __restrict__ hidden,
                                const float* __restrict__ Wout,
                                const float* __restrict__ bout,
                                float* __restrict__ out)
{
    const int r = blockIdx.x * blockDim.x + threadIdx.x;
    float acc = bout[0];
    #pragma unroll 8
    for (int k = 0; k < 64; ++k) acc += hidden[(size_t)r * 64 + k] * Wout[k];
    out[r] = acc;
}

extern "C" void kernel_launch(void* const* d_in, const int* in_sizes, int n_in,
                              void* d_out, int out_size, void* d_ws, size_t ws_size,
                              hipStream_t stream)
{
    (void)in_sizes; (void)n_in; (void)out_size; (void)ws_size;
    const float* x    = (const float*)d_in[0];
    const float* Wih0 = (const float*)d_in[1];
    const float* Whh0 = (const float*)d_in[2];
    const float* bih0 = (const float*)d_in[3];
    const float* bhh0 = (const float*)d_in[4];
    const float* Wih1 = (const float*)d_in[5];
    const float* Whh1 = (const float*)d_in[6];
    const float* bih1 = (const float*)d_in[7];
    const float* bhh1 = (const float*)d_in[8];
    const float* Wt   = (const float*)d_in[9];
    const float* bt   = (const float*)d_in[10];
    const float* av   = (const float*)d_in[11];
    const float* Wfc  = (const float*)d_in[12];
    const float* bfc  = (const float*)d_in[13];
    const float* Wout = (const float*)d_in[14];
    const float* bout = (const float*)d_in[15];

    const int N = 8192, T = 60;

    char* wsp = (char*)d_ws;
    size_t off = 0;
    auto alloc = [&](size_t bytes) {
        void* p = wsp + off;
        off += (bytes + 255) & ~(size_t)255;
        return p;
    };
    unsigned short* out0 = (unsigned short*)alloc((size_t)N * T * 64 * 2); // layer0 h-seq, bf16
    float* last   = (float*)alloc((size_t)N * 64 * 4);
    float* hp     = (float*)alloc((size_t)N * 64 * 4);
    float* s1     = (float*)alloc((size_t)N * 4);
    float* s2     = (float*)alloc((size_t)N * 4);
    float* res    = (float*)alloc((size_t)N * 64 * 4);
    float* hidden = (float*)alloc((size_t)N * 64 * 4);

    lstm_layer_kernel<0, 1, 0><<<N / 64, 256, 0, stream>>>(
        x, nullptr, Wih0, Whh0, bih0, bhh0, out0, nullptr, T);
    lstm_layer_kernel<1, 0, 1><<<N / 64, 256, 0, stream>>>(
        nullptr, (const unsigned int*)out0, Wih1, Whh1, bih1, bhh1, nullptr, last, T);
    gemm64_kernel<0><<<N / 128, 256, 0, stream>>>(last, Wt, bt, hp);
    gat_scores_kernel<<<N / 256, 256, 0, stream>>>(hp, av, s1, s2);
    gat_attn_kernel<<<N / 64, 256, 0, stream>>>(last, s1, s2, res, N);
    gemm64_kernel<1><<<N / 128, 256, 0, stream>>>(res, Wfc, bfc, hidden);
    out_head_kernel<<<N / 256, 256, 0, stream>>>(hidden, Wout, bout, (float*)d_out);
}
